// MultiHeadSelfAttention_46136538694303
// MI455X (gfx1250) — matrix operand order repaired and checked by reading, not compiled, not run on hardware
//
#include <hip/hip_runtime.h>
#include <hip/hip_bf16.h>

// ---------------------------------------------------------------------------
// Problem constants
// ---------------------------------------------------------------------------
#define BATCH   4
#define IN_DIM  256
#define RED_DIM 128
#define HEADS   4
#define HD      32        // head_dim
#define HP      64        // pooled H
#define WP      64        // pooled W
#define NQ      4096      // HP*WP
#define HF      128       // full H
#define WF      128       // full W
#define NPIX    16384     // HF*WF
#define QKV_M   384       // 3*RED_DIM

typedef __bf16 bf16;
typedef __attribute__((ext_vector_type(16))) __bf16 v16bf;
typedef __attribute__((ext_vector_type(8)))  __bf16 v8bf;
typedef __attribute__((ext_vector_type(8)))  float  v8f;

// ---------------------------------------------------------------------------
// WMMA helpers (wave32, 16x16x32 bf16 -> f32)
// ---------------------------------------------------------------------------
__device__ __forceinline__ v8f wmma_bf16(v16bf a, v16bf b, v8f c) {
  // (neg_a, A, neg_b, B, c_mod, C, reuse_a, reuse_b)
  return __builtin_amdgcn_wmma_f32_16x16x32_bf16(false, a, false, b,
                                                 (short)0, c, false, false);
}

// A fragment: 16x32 bf16, row-major source (K contiguous), leading dim `ld`.
// ISA layout: lane row = lane&15; VGPR0-3 hold K = hi*8..hi*8+7 (pairs),
// VGPR4-7 hold K = 16+hi*8..16+hi*8+7.
__device__ __forceinline__ v16bf frag_a_rm(const bf16* tile, int ld) {
  const int lane = threadIdx.x & 31;
  const int row  = lane & 15;
  const int hi   = lane >> 4;
  const bf16* p0 = tile + row * ld + hi * 8;
  const bf16* p1 = tile + row * ld + 16 + hi * 8;
  v8bf lo = *(const v8bf*)p0;
  v8bf hh = *(const v8bf*)p1;
  v16bf a;
#pragma unroll
  for (int i = 0; i < 8; ++i) { a[i] = lo[i]; a[8 + i] = hh[i]; }
  return a;
}

// B fragment: 32x16 bf16, element (k,n) at base + n*ldn + k (K contiguous).
// ISA layout: lane col = lane&15; VGPR0-3 hold K = hi*8..hi*8+7,
// VGPR4-7 hold K = 16+hi*8..16+hi*8+7 (same order as A).
__device__ __forceinline__ v16bf frag_b_kc(const bf16* base, int ldn) {
  const int lane = threadIdx.x & 31;
  const int col  = lane & 15;
  const int hi   = lane >> 4;
  const bf16* p = base + (size_t)col * ldn + hi * 8;
  v8bf lo = *(const v8bf*)p;
  v8bf hh = *(const v8bf*)(p + 16);
  v16bf b;
#pragma unroll
  for (int i = 0; i < 8; ++i) { b[i] = lo[i]; b[8 + i] = hh[i]; }
  return b;
}

// Reductions across the 16 lanes of a half-wave (xor masks < 16 never cross
// the lane[4] boundary on wave32).
__device__ __forceinline__ float rowmax16(float v) {
#pragma unroll
  for (int m = 1; m < 16; m <<= 1) v = fmaxf(v, __shfl_xor(v, m, 32));
  return v;
}
__device__ __forceinline__ float rowsum16(float v) {
#pragma unroll
  for (int m = 1; m < 16; m <<= 1) v += __shfl_xor(v, m, 32);
  return v;
}

// Online-softmax update for one 16x32 score tile (s0 = keys 0-15, s1 = 16-31).
// Rescales o-accumulators, updates m/l, writes bf16 P tile into LDS (ld=32).
__device__ __forceinline__ void softmax_update(v8f& s0, v8f& s1,
                                               float* m, float* l,
                                               v8f& o0, v8f& o1,
                                               bf16* ldsTile,
                                               int col, int hi) {
#pragma unroll
  for (int r = 0; r < 8; ++r) {
    float mx = rowmax16(fmaxf(s0[r], s1[r]));
    float mn = fmaxf(m[r], mx);
    float corr = __expf(m[r] - mn);
    float p0 = __expf(s0[r] - mn);
    float p1 = __expf(s1[r] - mn);
    l[r] = l[r] * corr + rowsum16(p0 + p1);
    m[r] = mn;
    o0[r] *= corr;
    o1[r] *= corr;
    ldsTile[(r + 8 * hi) * 32 + col]      = (bf16)p0;
    ldsTile[(r + 8 * hi) * 32 + 16 + col] = (bf16)p1;
  }
}

// ---------------------------------------------------------------------------
// 1) AvgPool 2x2 -> xd_t[b][pix][c] (bf16, channels contiguous)
// ---------------------------------------------------------------------------
__global__ __launch_bounds__(256) void pool_kernel(const float* __restrict__ x,
                                                   bf16* __restrict__ xdt) {
  size_t idx = (size_t)blockIdx.x * blockDim.x + threadIdx.x;
  if (idx >= (size_t)BATCH * NQ * IN_DIM) return;
  int c   = (int)(idx & (IN_DIM - 1));
  int pix = (int)((idx >> 8) & (NQ - 1));
  int b   = (int)(idx >> 20);
  int py = pix >> 6, px = pix & 63;
  const float* xp = x + (((size_t)b * IN_DIM + c) * HF + 2 * py) * WF + 2 * px;
  float v = 0.25f * (xp[0] + xp[1] + xp[WF] + xp[WF + 1]);
  xdt[idx] = (bf16)v;
}

// ---------------------------------------------------------------------------
// 2) f32 -> bf16 weight conversion
// ---------------------------------------------------------------------------
__global__ __launch_bounds__(256) void cvt_kernel(const float* __restrict__ s,
                                                  bf16* __restrict__ d, int n) {
  int i = blockIdx.x * blockDim.x + threadIdx.x;
  if (i < n) d[i] = (bf16)s[i];
}

// ---------------------------------------------------------------------------
// 3) QKV projection GEMM: [384x256] x [256 x 4096] per batch, WMMA bf16.
//    Epilogue: +bias, scale Q by 1/sqrt(32), scatter to
//      Q  [b][h][n][d]  (row-major, K-contig for score A-frags)
//      K  [b][h][n][d]  (d contiguous -> K-contig B-frags for Q*K^T)
//      Vt [b][h][d][n]  (n contiguous -> K-contig B-frags for P*V)
// ---------------------------------------------------------------------------
__global__ __launch_bounds__(128) void qkv_gemm(const bf16* __restrict__ wq,
                                                const bf16* __restrict__ xdt,
                                                const float* __restrict__ qkvb,
                                                bf16* __restrict__ Q,
                                                bf16* __restrict__ K,
                                                bf16* __restrict__ Vt) {
  const int wave = threadIdx.x >> 5;
  int tile = blockIdx.x * 4 + wave;          // 4 * 24 * 256 = 24576 tiles
  int b  = tile / (24 * 256);
  int t  = tile % (24 * 256);
  int mt = t >> 8;                           // 0..23
  int nt = t & 255;                          // 0..255 (pixel tile)
  const int lane = threadIdx.x & 31;
  const int col = lane & 15, hi = lane >> 4;

  const bf16* A  = wq + (size_t)mt * 16 * IN_DIM;
  const bf16* Bb = xdt + ((size_t)b * NQ + (size_t)nt * 16) * IN_DIM;
  v8f acc = {};
#pragma unroll
  for (int kc = 0; kc < IN_DIM; kc += 32) {
    v16bf a  = frag_a_rm(A + kc, IN_DIM);
    v16bf bb = frag_b_kc(Bb + kc, IN_DIM);
    acc = wmma_bf16(a, bb, acc);
  }
#pragma unroll
  for (int r = 0; r < 8; ++r) {
    int M = mt * 16 + r + 8 * hi;            // 0..383
    int n = nt * 16 + col;                   // pixel
    float val = acc[r] + qkvb[M];
    int s = M >> 7;                          // 0=Q 1=K 2=V
    int h = (M >> 5) & 3;
    int d = M & 31;
    size_t bh = (size_t)(b * HEADS + h);
    if (s == 0)      Q [(bh * NQ + n) * HD + d] = (bf16)(val * 0.17677669529663687f);
    else if (s == 1) K [(bh * NQ + n) * HD + d] = (bf16)val;
    else             Vt[(bh * HD + d) * NQ + n] = (bf16)val;
  }
}

// ---------------------------------------------------------------------------
// 4) Flash attention. One wave = 32 queries (two 16-row sub-tiles) of one
//    (b,h); streams 32 keys/iter. K and V B-fragments are loaded ONCE per
//    key block and reused by both query sub-tiles (2x arithmetic intensity).
//    Scores: wmma with Kdim = head_dim = 32 (exact fit). Softmax online f32.
//    P goes C-layout -> LDS -> A-layout. Output O[b][r=h*32+d][n] f32.
// ---------------------------------------------------------------------------
__global__ __launch_bounds__(128) void attn_kernel(const bf16* __restrict__ Q,
                                                   const bf16* __restrict__ K,
                                                   const bf16* __restrict__ Vt,
                                                   float* __restrict__ O) {
  const int wave = threadIdx.x >> 5;
  const int tile = blockIdx.x * 4 + wave;    // 4*4*128 = 2048 tiles (32 q each)
  const int b  = tile >> 9;                  // 512 tiles per batch
  const int h  = (tile >> 7) & 3;
  const int qt = tile & 127;                 // 32-query tile index
  const int lane = threadIdx.x & 31;
  const int col = lane & 15, hi = lane >> 4;

  __shared__ bf16 ldsP[4][2][16 * 32];       // 2KB per wave

  const size_t bh = (size_t)(b * HEADS + h);
  const bf16* Qbh = Q  + bh * NQ * HD;
  const bf16* Kbh = K  + bh * NQ * HD;
  const bf16* Vbh = Vt + bh * HD * NQ;

  const int qbase = qt * 32;
  const v16bf aq0 = frag_a_rm(Qbh + (size_t)qbase * HD, HD);
  const v16bf aq1 = frag_a_rm(Qbh + (size_t)(qbase + 16) * HD, HD);

  v8f o00 = {}, o01 = {};                    // qtile0: dims 0-15 / 16-31
  v8f o10 = {}, o11 = {};                    // qtile1
  float m0[8], l0[8], m1[8], l1[8];
#pragma unroll
  for (int r = 0; r < 8; ++r) {
    m0[r] = -3.0e38f; l0[r] = 0.0f;
    m1[r] = -3.0e38f; l1[r] = 0.0f;
  }

  for (int kb = 0; kb < NQ; kb += 32) {
    // --- shared K fragments (keys kb..kb+15 / kb+16..kb+31) ---
    v16bf k0 = frag_b_kc(Kbh + (size_t)kb * HD, HD);
    v16bf k1 = frag_b_kc(Kbh + (size_t)(kb + 16) * HD, HD);

    v8f s00 = {}, s01 = {}, s10 = {}, s11 = {};
    s00 = wmma_bf16(aq0, k0, s00);
    s01 = wmma_bf16(aq0, k1, s01);
    s10 = wmma_bf16(aq1, k0, s10);
    s11 = wmma_bf16(aq1, k1, s11);

    if (kb + 32 < NQ) {
      __builtin_prefetch(Kbh + (size_t)(kb + 32) * HD, 0, 1);
      __builtin_prefetch(Vbh + kb + 32, 0, 1);
    }

    softmax_update(s00, s01, m0, l0, o00, o01, &ldsP[wave][0][0], col, hi);
    softmax_update(s10, s11, m1, l1, o10, o11, &ldsP[wave][1][0], col, hi);

    __builtin_amdgcn_wave_barrier();
    asm volatile("s_wait_dscnt 0" ::: "memory");
    v16bf ap0 = frag_a_rm(&ldsP[wave][0][0], 32);
    v16bf ap1 = frag_a_rm(&ldsP[wave][1][0], 32);
    __builtin_amdgcn_wave_barrier();

    // --- shared V fragments (dims 0-15 / 16-31 over keys kb..kb+31) ---
    v16bf v0 = frag_b_kc(Vbh + kb, NQ);
    v16bf v1 = frag_b_kc(Vbh + (size_t)16 * NQ + kb, NQ);

    o00 = wmma_bf16(ap0, v0, o00);
    o01 = wmma_bf16(ap0, v1, o01);
    o10 = wmma_bf16(ap1, v0, o10);
    o11 = wmma_bf16(ap1, v1, o11);
  }

#pragma unroll
  for (int r = 0; r < 8; ++r) {
    float inv0 = 1.0f / l0[r];
    float inv1 = 1.0f / l1[r];
    int q0 = qbase + r + 8 * hi;
    int q1 = q0 + 16;
    size_t rowLo = (size_t)b * RED_DIM + h * HD + col;
    size_t rowHi = rowLo + 16;
    O[rowLo * NQ + q0] = o00[r] * inv0;
    O[rowHi * NQ + q0] = o01[r] * inv0;
    O[rowLo * NQ + q1] = o10[r] * inv1;
    O[rowHi * NQ + q1] = o11[r] * inv1;
  }
}

// ---------------------------------------------------------------------------
// 5) Bilinear 2x upsample (half-pixel centers) -> up_t[b][pix][r] bf16
// ---------------------------------------------------------------------------
__global__ __launch_bounds__(256) void upsample_kernel(const float* __restrict__ O,
                                                       bf16* __restrict__ upt) {
  size_t idx = (size_t)blockIdx.x * blockDim.x + threadIdx.x;
  if (idx >= (size_t)BATCH * NPIX * RED_DIM) return;
  int r   = (int)(idx & (RED_DIM - 1));
  int pix = (int)((idx >> 7) & (NPIX - 1));
  int b   = (int)(idx >> 21);
  int y = pix >> 7, xo = pix & 127;
  float iy = y * 0.5f - 0.25f;
  float ix = xo * 0.5f - 0.25f;
  int y0 = (int)floorf(iy); float wy = iy - (float)y0;
  int x0 = (int)floorf(ix); float wx = ix - (float)x0;
  int y0c = max(y0, 0), y1c = min(y0 + 1, HP - 1);
  int x0c = max(x0, 0), x1c = min(x0 + 1, WP - 1);
  const float* Op = O + ((size_t)b * RED_DIM + r) * NQ;
  float v00 = Op[y0c * WP + x0c], v01 = Op[y0c * WP + x1c];
  float v10 = Op[y1c * WP + x0c], v11 = Op[y1c * WP + x1c];
  float v = (1.0f - wy) * ((1.0f - wx) * v00 + wx * v01) +
            wy          * ((1.0f - wx) * v10 + wx * v11);
  upt[idx] = (bf16)v;
}

// ---------------------------------------------------------------------------
// 6) Output projection GEMM [256x128] x [128 x 16384] per batch + bias + x.
// ---------------------------------------------------------------------------
__global__ __launch_bounds__(128) void out_gemm(const bf16* __restrict__ wo,
                                                const bf16* __restrict__ upt,
                                                const float* __restrict__ ob,
                                                const float* __restrict__ x,
                                                float* __restrict__ out) {
  const int wave = threadIdx.x >> 5;
  int tile = blockIdx.x * 4 + wave;           // 4 * 16 * 1024 = 65536 tiles
  int b  = tile >> 14;
  int t  = tile & 16383;
  int mt = t >> 10;                           // 0..15
  int nt = t & 1023;                          // 0..1023 (pixel tile)
  const int lane = threadIdx.x & 31;
  const int col = lane & 15, hi = lane >> 4;

  const bf16* A  = wo + (size_t)mt * 16 * RED_DIM;
  const bf16* Bb = upt + ((size_t)b * NPIX + (size_t)nt * 16) * RED_DIM;
  v8f acc = {};
#pragma unroll
  for (int kc = 0; kc < RED_DIM; kc += 32) {
    v16bf a  = frag_a_rm(A + kc, RED_DIM);
    v16bf bb = frag_b_kc(Bb + kc, RED_DIM);
    acc = wmma_bf16(a, bb, acc);
  }
#pragma unroll
  for (int r = 0; r < 8; ++r) {
    int M = mt * 16 + r + 8 * hi;
    int pix = nt * 16 + col;
    size_t o = ((size_t)b * IN_DIM + M) * NPIX + pix;
    out[o] = acc[r] + ob[M] + x[o];
  }
}

// ---------------------------------------------------------------------------
// Launch
// ---------------------------------------------------------------------------
extern "C" void kernel_launch(void* const* d_in, const int* in_sizes, int n_in,
                              void* d_out, int out_size, void* d_ws, size_t ws_size,
                              hipStream_t stream) {
  const float* x     = (const float*)d_in[0];
  const float* qkv_w = (const float*)d_in[1];
  const float* qkv_b = (const float*)d_in[2];
  const float* out_w = (const float*)d_in[3];
  const float* out_b = (const float*)d_in[4];
  float* out = (float*)d_out;

  // Workspace carve (bytes), all 256B-aligned by construction:
  char* ws = (char*)d_ws;
  bf16* xdt = (bf16*)ws;                                   //  8 MB
  bf16* wq  = (bf16*)(ws + ((size_t)8 << 20));             //  192 KB
  bf16* wo  = (bf16*)(ws + ((size_t)8 << 20) + 196608);    //  64 KB
  bf16* Q   = (bf16*)(ws + ((size_t)8 << 20) + 262144);    //  4 MB
  bf16* K   = Q + (size_t)BATCH * HEADS * NQ * HD;         //  4 MB
  bf16* Vt  = K + (size_t)BATCH * HEADS * NQ * HD;         //  4 MB
  float* O  = (float*)(Vt + (size_t)BATCH * HEADS * NQ * HD); // 8 MB
  bf16* upt = (bf16*)(O + (size_t)BATCH * RED_DIM * NQ);   // 16 MB

  pool_kernel<<<(BATCH * NQ * IN_DIM) / 256, 256, 0, stream>>>(x, xdt);
  cvt_kernel<<<(QKV_M * IN_DIM + 255) / 256, 256, 0, stream>>>(qkv_w, wq, QKV_M * IN_DIM);
  cvt_kernel<<<(IN_DIM * RED_DIM + 255) / 256, 256, 0, stream>>>(out_w, wo, IN_DIM * RED_DIM);
  qkv_gemm<<<(BATCH * 24 * 256) / 4, 128, 0, stream>>>(wq, xdt, qkv_b, Q, K, Vt);
  attn_kernel<<<(BATCH * HEADS * (NQ / 32)) / 4, 128, 0, stream>>>(Q, K, Vt, O);
  upsample_kernel<<<((size_t)BATCH * NPIX * RED_DIM) / 256, 256, 0, stream>>>(O, upt);
  out_gemm<<<(BATCH * 16 * 1024) / 4, 128, 0, stream>>>(wo, upt, out_b, x, out);
}
